// EvalRoutedQuantizedMoE_5205500362821
// MI455X (gfx1250) — matrix operand order repaired and checked by reading, not compiled, not run on hardware
//
#include <hip/hip_runtime.h>
#include <hip/hip_bf16.h>

// Problem constants (fixed by reference setup_inputs)
#define TOK 2048   // B*S
#define DIM 1024   // hidden
#define DFS 2048   // shared ffn
#define DFE 512    // expert ffn
#define NE  8      // experts (top_k = 2)

// LDS B-tile: 64 rows (N) x 64 halves (K), row stride padded to 72 halves
// (144 B = 36 banks) -> the 16 rows a fragment touches land on 16 disjoint
// 4-bank groups: conflict-free ds_load_b128.
#define BROW 72
#define BTILE_HALFS (64*BROW)   // 4608 halves = 9216 B per buffer

typedef __attribute__((ext_vector_type(16))) _Float16 v16h;
typedef __attribute__((ext_vector_type(8)))  _Float16 v8h;
typedef __attribute__((ext_vector_type(8)))  float    v8f;

__device__ __forceinline__ float silu_f(float v){ return v / (1.0f + __expf(-v)); }

__device__ __forceinline__ v16h pack16(v8h lo, v8h hi){
  v16h r;
#pragma unroll
  for (int i = 0; i < 8; i++){ r[i] = lo[i]; r[i+8] = hi[i]; }
  return r;
}

// A fragment (16x32 f16). rowbase = &A[row_of_this_lane][0].
// lanes 0-15: K=k..k+7 and k+16..k+23 ; lanes 16-31: K=k+8..15 and k+24..31.
__device__ __forceinline__ v16h load_a(const _Float16* __restrict__ rowbase, int k, int sel){
  const _Float16* p = rowbase + k + sel*8;
  v8h lo = *(const v8h*)(p);
  v8h hi = *(const v8h*)(p + 16);
  return pack16(lo, hi);
}

// B fragment from the LDS-staged tile (row = W row = B column, contiguous K).
__device__ __forceinline__ v16h ldsb_frag(const _Float16* lbuf, int s, int kk, int lane){
  int ln = lane & 15, sel = lane >> 4;
  const _Float16* p = lbuf + (16*s + ln)*BROW + kk + sel*8;
  v8h lo = *(const v8h*)(p);
  v8h hi = *(const v8h*)(p + 16);
  return pack16(lo, hi);
}

#define WMMA_F16(a,b,c) __builtin_amdgcn_wmma_f32_16x16x32_f16(false,(a),false,(b),(short)0,(c),false,false)

// ---- CDNA5 async global->LDS copy (ASYNCcnt-tracked), 16 B per lane --------
// Inline asm (portable across toolchains; clang builtin prototypes differ).
__device__ __forceinline__ void async_g2l_16B(const _Float16* g, _Float16* l){
  unsigned loff = (unsigned)(unsigned long long)(__attribute__((address_space(3))) _Float16*)l;
  asm volatile("global_load_async_to_lds_b128 %0, %1, off"
               :: "v"(loff), "v"(g) : "memory");
}

template<int N> __device__ __forceinline__ void wait_async(){
#if __has_builtin(__builtin_amdgcn_s_wait_asynccnt)
  __builtin_amdgcn_s_wait_asynccnt((unsigned short)N);
#else
  asm volatile("s_wait_asynccnt %0" :: "i"(N) : "memory");
#endif
}

// Stage a 64(N) x 64(K) f16 tile of W (row-major [N][ldk]) into LDS.
// 512 x 16B chunks over 128 threads -> 4 async instructions per wave.
template<int LDK>
__device__ __forceinline__ void stage_b_tile(const _Float16* __restrict__ W, int n0, int k,
                                             _Float16* lbuf, int tid){
#pragma unroll
  for (int i = 0; i < 4; i++){
    int c = tid + i*128;          // 0..511
    int row = c >> 3, part = c & 7;
    const _Float16* g = W + (size_t)(n0 + row)*LDK + k + part*8;
    _Float16* l = lbuf + row*BROW + part*8;
    async_g2l_16B(g, l);
  }
}

// ---------------------------------------------------------------- converters
__global__ void cvt_f32_to_f16_kernel(const float* __restrict__ s, _Float16* __restrict__ d, int n){
  int i = blockIdx.x*blockDim.x + threadIdx.x;
  if (i < n) d[i] = (_Float16)s[i];
}
__global__ void cvt_i32_to_f16_kernel(const int* __restrict__ s, _Float16* __restrict__ d, int n){
  int i = blockIdx.x*blockDim.x + threadIdx.x;
  if (i < n) d[i] = (_Float16)(float)s[i];  // |q| <= 128 -> exact in f16
}

// ---------------------------------------------------------------- router
__global__ __launch_bounds__(128) void moe_router_kernel(
    const float* __restrict__ x, const float* __restrict__ rw,
    const float* __restrict__ alpha,
    _Float16* __restrict__ xh, int* __restrict__ e0a, int* __restrict__ e1a,
    float* __restrict__ wa0, float* __restrict__ wa1, float* __restrict__ cfac)
{
  int t = blockIdx.x;
  int tid = threadIdx.x;
  const float* xr = x + (size_t)t*DIM;
  float acc[NE];
#pragma unroll
  for (int e = 0; e < NE; e++) acc[e] = 0.f;
  for (int d = tid; d < DIM; d += 128){
    float xv = xr[d];
    xh[(size_t)t*DIM + d] = (_Float16)xv;
#pragma unroll
    for (int e = 0; e < NE; e++) acc[e] += xv * rw[e*DIM + d];
  }
  __shared__ float red[128*NE];
#pragma unroll
  for (int e = 0; e < NE; e++) red[tid*NE + e] = acc[e];
  __syncthreads();
  for (int off = 64; off > 0; off >>= 1){
    if (tid < off){
#pragma unroll
      for (int e = 0; e < NE; e++) red[tid*NE + e] += red[(tid+off)*NE + e];
    }
    __syncthreads();
  }
  if (tid == 0){
    float l0 = -3.4e38f; int e0 = 0;
    for (int e = 0; e < NE; e++){ float v = red[e]; if (v > l0){ l0 = v; e0 = e; } }
    float l1 = -3.4e38f; int e1 = (e0 == 0) ? 1 : 0;
    for (int e = 0; e < NE; e++){ if (e == e0) continue; float v = red[e]; if (v > l1){ l1 = v; e1 = e; } }
    float z   = __expf(l1 - l0);
    float inv = 1.0f / (1.0f + z);
    float a0 = inv     * alpha[e0];
    float a1 = z * inv * alpha[e1];
    e0a[t] = e0; e1a[t] = e1; wa0[t] = a0; wa1[t] = a1;
    cfac[t] = 1.0f - (a0 + a1);
  }
}

// ---------------------------------------------------------------- list build
__global__ void moe_build_lists_kernel(
    const int* __restrict__ e0a, const int* __restrict__ e1a,
    const float* __restrict__ wa0, const float* __restrict__ wa1,
    int* __restrict__ counts, int* __restrict__ perm,
    float* __restrict__ pwv, int* __restrict__ slot)
{
  int e = blockIdx.x;
  int lane = threadIdx.x;
  int cnt = 0;
  for (int base = 0; base < TOK; base += 32){
    int t = base + lane;
    bool s0 = (e0a[t] == e);
    bool s1 = (e1a[t] == e);
    bool hit = s0 || s1;
    unsigned long long m = __ballot(hit);
    int pre = (int)__popcll(m & ((1ull << lane) - 1ull));
    if (hit){
      int i = cnt + pre;
      perm[e*TOK + i] = t;
      pwv [e*TOK + i] = s0 ? wa0[t] : wa1[t];
      slot[e*TOK + i] = s0 ? 0 : 1;
    }
    cnt += (int)__popcll(m);
  }
  if (lane == 0) counts[e] = cnt;
}

// ---------------------------------------------------------------- shared MLP
// hsh = silu(xh @ gw^T) * (xh @ uw^T), f16 [TOK, DFS]
__global__ __launch_bounds__(128) void moe_shared_gateup_kernel(
    const _Float16* __restrict__ xh, const _Float16* __restrict__ gw,
    const _Float16* __restrict__ uw, _Float16* __restrict__ hsh)
{
  __shared__ _Float16 lg[2][BTILE_HALFS];
  __shared__ _Float16 lu[2][BTILE_HALFS];
  int tid = threadIdx.x, lane = tid & 31, wave = tid >> 5;
  int m0 = blockIdx.x*64 + wave*16;
  int n0 = blockIdx.y*64;
  int sel = lane >> 4;
  const _Float16* arow = xh + (size_t)(m0 + (lane & 15))*DIM;
  v8f g[4] = {}, u[4] = {};
  constexpr int NIT = DIM/64;
  stage_b_tile<DIM>(gw, n0, 0, &lg[0][0], tid);
  stage_b_tile<DIM>(uw, n0, 0, &lu[0][0], tid);
  for (int it = 0; it < NIT; ++it){
    int k = it*64;
    if (it + 1 < NIT){
      stage_b_tile<DIM>(gw, n0, k+64, &lg[(it+1)&1][0], tid);
      stage_b_tile<DIM>(uw, n0, k+64, &lu[(it+1)&1][0], tid);
      wait_async<8>();
    } else {
      wait_async<0>();
    }
    __syncthreads();
    const _Float16* bg = &lg[it&1][0];
    const _Float16* bu = &lu[it&1][0];
#pragma unroll
    for (int kk = 0; kk < 64; kk += 32){
      v16h a = load_a(arow, k+kk, sel);
#pragma unroll
      for (int s = 0; s < 4; s++){
        g[s] = WMMA_F16(a, ldsb_frag(bg, s, kk, lane), g[s]);
        u[s] = WMMA_F16(a, ldsb_frag(bu, s, kk, lane), u[s]);
      }
    }
    __syncthreads();
  }
  int ln = lane & 15;
#pragma unroll
  for (int s = 0; s < 4; s++){
    int n = n0 + 16*s + ln;
#pragma unroll
    for (int j = 0; j < 8; j++){
      int m = m0 + j + 8*sel;
      hsh[(size_t)m*DFS + n] = (_Float16)(silu_f(g[s][j]) * u[s][j]);
    }
  }
}

// out = (hsh @ dw^T) * cfac[t]   (the "(1 - sum w*alpha)*shared" term)
__global__ __launch_bounds__(128) void moe_shared_down_kernel(
    const _Float16* __restrict__ hsh, const _Float16* __restrict__ dw,
    const float* __restrict__ cfac, float* __restrict__ out)
{
  __shared__ _Float16 lb[2][BTILE_HALFS];
  int tid = threadIdx.x, lane = tid & 31, wave = tid >> 5;
  int m0 = blockIdx.x*64 + wave*16;
  int n0 = blockIdx.y*64;
  int sel = lane >> 4;
  const _Float16* arow = hsh + (size_t)(m0 + (lane & 15))*DFS;
  v8f c[4] = {};
  constexpr int NIT = DFS/64;
  stage_b_tile<DFS>(dw, n0, 0, &lb[0][0], tid);
  for (int it = 0; it < NIT; ++it){
    int k = it*64;
    if (it + 1 < NIT){
      stage_b_tile<DFS>(dw, n0, k+64, &lb[(it+1)&1][0], tid);
      wait_async<4>();
    } else {
      wait_async<0>();
    }
    __syncthreads();
    const _Float16* bb = &lb[it&1][0];
#pragma unroll
    for (int kk = 0; kk < 64; kk += 32){
      v16h a = load_a(arow, k+kk, sel);
#pragma unroll
      for (int s = 0; s < 4; s++)
        c[s] = WMMA_F16(a, ldsb_frag(bb, s, kk, lane), c[s]);
    }
    __syncthreads();
  }
  int ln = lane & 15;
  float cf[8];
#pragma unroll
  for (int j = 0; j < 8; j++) cf[j] = cfac[m0 + j + 8*sel];
#pragma unroll
  for (int s = 0; s < 4; s++){
    int n = n0 + 16*s + ln;
#pragma unroll
    for (int j = 0; j < 8; j++){
      int m = m0 + j + 8*sel;
      out[(size_t)m*DIM + n] = c[s][j] * cf[j];
    }
  }
}

// ---------------------------------------------------------------- experts
__global__ __launch_bounds__(128) void moe_expert_gateup_kernel(
    const _Float16* __restrict__ xh, const _Float16* __restrict__ gqh,
    const _Float16* __restrict__ uqh, const float* __restrict__ gs,
    const float* __restrict__ us, const int* __restrict__ counts,
    const int* __restrict__ perm, _Float16* __restrict__ h2)
{
  __shared__ _Float16 lg[2][BTILE_HALFS];
  __shared__ _Float16 lu[2][BTILE_HALFS];
  int e = blockIdx.z;
  int nE = counts[e];
  int m0b = blockIdx.x*64;
  if (m0b >= nE) return;                 // block-uniform: no barrier divergence
  int tid = threadIdx.x, lane = tid & 31, wave = tid >> 5;
  int m0 = m0b + wave*16;
  int n0 = blockIdx.y*64;
  int sel = lane >> 4;
  int mi = m0 + (lane & 15);
  int mc = mi < nE ? mi : nE - 1;        // clamp gather; stores guarded below
  int tok = perm[e*TOK + mc];
  const _Float16* arow = xh + (size_t)tok*DIM;
  const _Float16* gw = gqh + (size_t)e*DFE*DIM;
  const _Float16* uw = uqh + (size_t)e*DFE*DIM;
  v8f g[4] = {}, u[4] = {};
  constexpr int NIT = DIM/64;
  stage_b_tile<DIM>(gw, n0, 0, &lg[0][0], tid);
  stage_b_tile<DIM>(uw, n0, 0, &lu[0][0], tid);
  for (int it = 0; it < NIT; ++it){
    int k = it*64;
    if (it + 1 < NIT){
      stage_b_tile<DIM>(gw, n0, k+64, &lg[(it+1)&1][0], tid);
      stage_b_tile<DIM>(uw, n0, k+64, &lu[(it+1)&1][0], tid);
      wait_async<8>();
    } else {
      wait_async<0>();
    }
    __syncthreads();
    const _Float16* bg = &lg[it&1][0];
    const _Float16* bu = &lu[it&1][0];
#pragma unroll
    for (int kk = 0; kk < 64; kk += 32){
      v16h a = load_a(arow, k+kk, sel);
#pragma unroll
      for (int s = 0; s < 4; s++){
        g[s] = WMMA_F16(a, ldsb_frag(bg, s, kk, lane), g[s]);
        u[s] = WMMA_F16(a, ldsb_frag(bu, s, kk, lane), u[s]);
      }
    }
    __syncthreads();
  }
  int ln = lane & 15;
#pragma unroll
  for (int s = 0; s < 4; s++){
    int n = n0 + 16*s + ln;
    float sg = gs[e*DFE + n], su = us[e*DFE + n];
#pragma unroll
    for (int j = 0; j < 8; j++){
      int m = m0 + j + 8*sel;
      if (m < nE){
        float hv = silu_f(g[s][j]*sg) * (u[s][j]*su);
        h2[((size_t)e*TOK + m)*DFE + n] = (_Float16)hv;
      }
    }
  }
}

__global__ __launch_bounds__(128) void moe_expert_down_kernel(
    const _Float16* __restrict__ h2, const _Float16* __restrict__ dqh,
    const float* __restrict__ dsc, const int* __restrict__ counts,
    const int* __restrict__ perm, const float* __restrict__ pwv,
    const int* __restrict__ slot, float* __restrict__ contrib)
{
  __shared__ _Float16 lb[2][BTILE_HALFS];
  int e = blockIdx.z;
  int nE = counts[e];
  int m0b = blockIdx.x*64;
  if (m0b >= nE) return;
  int tid = threadIdx.x, lane = tid & 31, wave = tid >> 5;
  int m0 = m0b + wave*16;
  int n0 = blockIdx.y*64;
  int sel = lane >> 4;
  const _Float16* A = h2 + (size_t)e*TOK*DFE;
  const _Float16* arow = A + (size_t)(m0 + (lane & 15))*DFE;
  const _Float16* W = dqh + (size_t)e*DIM*DFE;
  v8f c[4] = {};
  constexpr int NIT = DFE/64;
  stage_b_tile<DFE>(W, n0, 0, &lb[0][0], tid);
  for (int it = 0; it < NIT; ++it){
    int k = it*64;
    if (it + 1 < NIT){
      stage_b_tile<DFE>(W, n0, k+64, &lb[(it+1)&1][0], tid);
      wait_async<4>();
    } else {
      wait_async<0>();
    }
    __syncthreads();
    const _Float16* bb = &lb[it&1][0];
#pragma unroll
    for (int kk = 0; kk < 64; kk += 32){
      v16h a = load_a(arow, k+kk, sel);
#pragma unroll
      for (int s = 0; s < 4; s++)
        c[s] = WMMA_F16(a, ldsb_frag(bb, s, kk, lane), c[s]);
    }
    __syncthreads();
  }
  int ln = lane & 15;
  int tokv[8]; float wv[8]; int slv[8];
#pragma unroll
  for (int j = 0; j < 8; j++){
    int m = m0 + j + 8*sel;
    int mc = m < nE ? m : nE - 1;
    tokv[j] = perm[e*TOK + mc];
    wv[j]   = pwv [e*TOK + mc];
    slv[j]  = slot[e*TOK + mc];
  }
#pragma unroll
  for (int s = 0; s < 4; s++){
    int n = n0 + 16*s + ln;
    float sd = dsc[e*DIM + n];
#pragma unroll
    for (int j = 0; j < 8; j++){
      int m = m0 + j + 8*sel;
      if (m < nE){
        contrib[((size_t)slv[j]*TOK + tokv[j])*DIM + n] = c[s][j] * sd * wv[j];
      }
    }
  }
}

// out += contrib[slot0] + contrib[slot1], fixed order -> deterministic
__global__ void moe_combine_kernel(float* __restrict__ out, const float* __restrict__ contrib){
  int i = blockIdx.x*blockDim.x + threadIdx.x;
  if (i < TOK*DIM) out[i] += contrib[i] + contrib[(size_t)TOK*DIM + i];
}

// ---------------------------------------------------------------- launch
extern "C" void kernel_launch(void* const* d_in, const int* in_sizes, int n_in,
                              void* d_out, int out_size, void* d_ws, size_t ws_size,
                              hipStream_t stream) {
  const float* x     = (const float*)d_in[0];
  const float* rw    = (const float*)d_in[1];
  const float* sgw   = (const float*)d_in[2];
  const float* suw   = (const float*)d_in[3];
  const float* sdw   = (const float*)d_in[4];
  const float* gs    = (const float*)d_in[5];
  const float* us    = (const float*)d_in[6];
  const float* dsc   = (const float*)d_in[7];
  const float* alpha = (const float*)d_in[8];
  const int*   gq    = (const int*)d_in[9];
  const int*   uq    = (const int*)d_in[10];
  const int*   dq    = (const int*)d_in[11];
  float* out = (float*)d_out;
  (void)in_sizes; (void)n_in; (void)out_size; (void)ws_size;

  char* ws = (char*)d_ws;
  size_t off = 0;
  auto carve = [&](size_t bytes) -> char* {
    char* p = ws + off;
    off += (bytes + 255) & ~(size_t)255;
    return p;
  };
  _Float16* xh   = (_Float16*)carve((size_t)TOK*DIM*2);
  _Float16* gwh  = (_Float16*)carve((size_t)DFS*DIM*2);
  _Float16* uwh  = (_Float16*)carve((size_t)DFS*DIM*2);
  _Float16* dwh  = (_Float16*)carve((size_t)DIM*DFS*2);
  _Float16* gqh  = (_Float16*)carve((size_t)NE*DFE*DIM*2);
  _Float16* uqh  = (_Float16*)carve((size_t)NE*DFE*DIM*2);
  _Float16* dqh  = (_Float16*)carve((size_t)NE*DIM*DFE*2);
  _Float16* hsh  = (_Float16*)carve((size_t)TOK*DFS*2);
  _Float16* h2   = (_Float16*)carve((size_t)NE*TOK*DFE*2);
  float*    contrib = (float*)carve((size_t)2*TOK*DIM*4);
  int*   e0a  = (int*)  carve((size_t)TOK*4);
  int*   e1a  = (int*)  carve((size_t)TOK*4);
  float* wa0  = (float*)carve((size_t)TOK*4);
  float* wa1  = (float*)carve((size_t)TOK*4);
  float* cfac = (float*)carve((size_t)TOK*4);
  int*   counts = (int*)carve((size_t)NE*4);
  int*   perm = (int*)  carve((size_t)NE*TOK*4);
  float* pwv  = (float*)carve((size_t)NE*TOK*4);
  int*   slotb= (int*)  carve((size_t)NE*TOK*4);

  cvt_f32_to_f16_kernel<<<(DFS*DIM+255)/256, 256, 0, stream>>>(sgw, gwh, DFS*DIM);
  cvt_f32_to_f16_kernel<<<(DFS*DIM+255)/256, 256, 0, stream>>>(suw, uwh, DFS*DIM);
  cvt_f32_to_f16_kernel<<<(DIM*DFS+255)/256, 256, 0, stream>>>(sdw, dwh, DIM*DFS);
  cvt_i32_to_f16_kernel<<<(NE*DFE*DIM+255)/256, 256, 0, stream>>>(gq, gqh, NE*DFE*DIM);
  cvt_i32_to_f16_kernel<<<(NE*DFE*DIM+255)/256, 256, 0, stream>>>(uq, uqh, NE*DFE*DIM);
  cvt_i32_to_f16_kernel<<<(NE*DIM*DFE+255)/256, 256, 0, stream>>>(dq, dqh, NE*DIM*DFE);

  moe_router_kernel<<<TOK, 128, 0, stream>>>(x, rw, alpha, xh, e0a, e1a, wa0, wa1, cfac);
  moe_build_lists_kernel<<<NE, 32, 0, stream>>>(e0a, e1a, wa0, wa1, counts, perm, pwv, slotb);
  moe_shared_gateup_kernel<<<dim3(TOK/64, DFS/64), 128, 0, stream>>>(xh, gwh, uwh, hsh);
  moe_shared_down_kernel<<<dim3(TOK/64, DIM/64), 128, 0, stream>>>(hsh, dwh, cfac, out);
  moe_expert_gateup_kernel<<<dim3(TOK/64, DFE/64, NE), 128, 0, stream>>>(xh, gqh, uqh, gs, us, counts, perm, h2);
  moe_expert_down_kernel<<<dim3(TOK/64, DIM/64, NE), 128, 0, stream>>>(h2, dqh, dsc, counts, perm, pwv, slotb, contrib);
  moe_combine_kernel<<<(TOK*DIM+255)/256, 256, 0, stream>>>(out, contrib);
}
